// CrystalDynamics_23063974379745
// MI455X (gfx1250) — matrix operand order repaired and checked by reading, not compiled, not run on hardware
//
#include <hip/hip_runtime.h>
#include <hip/hip_bf16.h>
#include <math.h>

#define BB   512        // crystals
#define NPC  32         // atoms per crystal
#define NN   (BB*NPC)   // 16384 nodes
#define KNN  12
#define EE   (NN*KNN)   // 196608 edges
#define DD   64
#define TT   64
#define LL   4

typedef __attribute__((ext_vector_type(16))) _Float16 v16h;
typedef __attribute__((ext_vector_type(8)))  _Float16 v8h;
typedef __attribute__((ext_vector_type(8)))  float    v8f;

// ---------------- WMMA fragment helpers (CDNA5 16x16x32 f16 layouts) ----------------
// A 16x32 f16: lane m = L%16; lanes 0-15 hold K=0..7 & 16..23, lanes 16-31 K=8..15 & 24..31
__device__ __forceinline__ v16h ldA(const _Float16* base, int row0, int stride, int kk, int lane) {
    int m  = lane & 15;
    int kb = (lane & 16) ? 8 : 0;
    const _Float16* p = base + (row0 + m) * stride + kk * 32 + kb;
    v8h lo = *(const v8h*)(p);
    v8h hi = *(const v8h*)(p + 16);
    v16h a;
#pragma unroll
    for (int i = 0; i < 8; ++i) { a[i] = lo[i]; a[8 + i] = hi[i]; }
    return a;
}
// B 32x16 f16 from transposed weights wt[n][Kr]: lane n = L%16; lanes 0-15 K=0..7 & 16..23, lanes 16-31 K=8..15 & 24..31
__device__ __forceinline__ v16h ldB(const _Float16* wt, int Kr, int n0, int krow0, int lane) {
    int n  = lane & 15;
    int kb = (lane & 16) ? 8 : 0;
    const _Float16* p = wt + (n0 + n) * Kr + krow0 + kb;
    v8h lo = *(const v8h*)(p);
    v8h hi = *(const v8h*)(p + 16);
    v16h b;
#pragma unroll
    for (int i = 0; i < 8; ++i) { b[i] = lo[i]; b[8 + i] = hi[i]; }
    return b;
}
__device__ __forceinline__ v8f wmma32(v16h a, v16h b, v8f c) {
    return __builtin_amdgcn_wmma_f32_16x16x32_f16(false, a, false, b, (short)0, c, false, false);
}
__device__ __forceinline__ float silu_f(float x) { return x / (1.f + __expf(-x)); }

// CDNA5 async memory->LDS copy: 16B per lane, tracked by ASYNCcnt
__device__ __forceinline__ void async_g2l_b128(unsigned lds_off, const void* gaddr) {
    asm volatile("global_load_async_to_lds_b128 %0, %1, off"
                 :: "v"(lds_off), "v"((unsigned long long)(size_t)gaddr) : "memory");
}
__device__ __forceinline__ void wait_async0() {
    asm volatile("s_wait_asynccnt 0x0" ::: "memory");
}
__device__ __forceinline__ unsigned lds_off_of(const void* p) {
    return (unsigned)(size_t)p;   // LDS aperture: addr[31:0] is the LDS byte address
}

// GEMM stage: [96,64] x [64,64] -> silu -> [96,64] f16 (24 tiles over 8 waves)
__device__ __forceinline__ void gemm_silu_96(const _Float16* In, _Float16* Out,
                                             const _Float16* wt, const float* bias,
                                             int wave, int lane) {
    for (int t = wave; t < 24; t += 8) {
        int row0 = (t >> 2) << 4, n0 = (t & 3) << 4;
        v8f acc = {};
#pragma unroll
        for (int ks = 0; ks < 2; ++ks)
            acc = wmma32(ldA(In, row0, 64, ks, lane), ldB(wt, 64, n0, ks * 32, lane), acc);
        int n = lane & 15, mo = (lane & 16) ? 8 : 0;
        float bv = bias[n0 + n];
#pragma unroll
        for (int r = 0; r < 8; ++r) {
            int m = row0 + mo + r;
            Out[m * 64 + n0 + n] = (_Float16)silu_f(acc[r] + bv);
        }
    }
}

// ---------------- small utility kernels ----------------
__global__ void k_f16t(const float* __restrict__ src, _Float16* __restrict__ dst,
                       int L_, int Kr, int KrPad) {
    int i = blockIdx.x * blockDim.x + threadIdx.x;
    int total = L_ * 64 * KrPad;
    if (i >= total) return;
    int k = i % KrPad; int rem = i / KrPad; int n = rem % 64; int l = rem / 64;
    float v = (k < Kr) ? src[(l * Kr + k) * 64 + n] : 0.f;
    dst[i] = (_Float16)v;
}

__global__ void k_zero(float* p, int n) {
    int i = blockIdx.x * blockDim.x + threadIdx.x;
    if (i < n) p[i] = 0.f;
}

// time embedding: sinusoid -> Linear(64,128) -> SiLU -> Linear(128,64)
__global__ void k_time(const float* __restrict__ t, const float* __restrict__ w1,
                       const float* __restrict__ b1, const float* __restrict__ w2,
                       const float* __restrict__ b2, float* __restrict__ tfeat) {
    __shared__ float ssin[64];
    __shared__ float shid[128];
    int b = blockIdx.x, c = threadIdx.x;  // 64 threads
    float tv = t[b];
    float neg = -logf(10000.f) / 31.f;
    if (c < 32) ssin[c] = __sinf(tv * __expf(neg * c));
    else        ssin[c] = __cosf(tv * __expf(neg * (c - 32)));
    __syncthreads();
#pragma unroll
    for (int rep = 0; rep < 2; ++rep) {
        int j = c + rep * 64;
        float s = b1[j];
        for (int i = 0; i < 64; ++i) s += ssin[i] * w1[i * 128 + j];
        shid[j] = silu_f(s);
    }
    __syncthreads();
    float s = b2[c];
    for (int j = 0; j < 128; ++j) s += shid[j] * w2[j * 64 + c];
    tfeat[b * 64 + c] = s;
}

// per-(layer, crystal) projections of t_feat through the tail rows of edge_w1/node_w1 (+bias)
__global__ void k_tproj(const float* __restrict__ tfeat, const float* __restrict__ ew1,
                        const float* __restrict__ eb1, const float* __restrict__ nw1,
                        const float* __restrict__ nb1, float* __restrict__ tpe,
                        float* __restrict__ tpn) {
    int bid = blockIdx.x;           // L*B blocks
    int l = bid / BB, b = bid % BB;
    __shared__ float st[64];
    int c = threadIdx.x;            // 64 threads
    st[c] = tfeat[b * 64 + c];
    __syncthreads();
    float se = eb1[l * 64 + c];
    float sn = nb1[l * 64 + c];
    for (int k = 0; k < 64; ++k) {
        float tv = st[k];
        se += tv * ew1[(l * 193 + 129 + k) * 64 + c];
        sn += tv * nw1[(l * 192 + 128 + k) * 64 + c];
    }
    tpe[(l * BB + b) * 64 + c] = se;
    tpn[(l * BB + b) * 64 + c] = sn;
}

// PBC KNN: one wave per crystal, lane = atom; mask-based top-12 via recomputation
__global__ void k_knn(const float* __restrict__ frac, const float* __restrict__ lat,
                      int* __restrict__ idxg, float* __restrict__ dcart,
                      float* __restrict__ dsqg) {
    __shared__ float sf[NPC][3];
    __shared__ float sl[9];
    int cry = blockIdx.x, i = threadIdx.x;  // 32 threads
#pragma unroll
    for (int d = 0; d < 3; ++d) sf[i][d] = frac[(cry * NPC + i) * 3 + d];
    if (i < 9) sl[i] = lat[cry * 9 + i];
    __syncthreads();
    float f0 = sf[i][0], f1 = sf[i][1], f2 = sf[i][2];
    unsigned mask = 1u << i;  // exclude self
    for (int k = 0; k < KNN; ++k) {
        float best = 1e30f; int bj = 0;
        float b0 = 0.f, b1 = 0.f, b2 = 0.f;
        for (int j = 0; j < NPC; ++j) {
            if (mask & (1u << j)) continue;
            float d0 = f0 - sf[j][0]; d0 -= rintf(d0);
            float d1 = f1 - sf[j][1]; d1 -= rintf(d1);
            float d2 = f2 - sf[j][2]; d2 -= rintf(d2);
            float c0 = d0 * sl[0] + d1 * sl[3] + d2 * sl[6];
            float c1 = d0 * sl[1] + d1 * sl[4] + d2 * sl[7];
            float c2 = d0 * sl[2] + d1 * sl[5] + d2 * sl[8];
            float ds = c0 * c0 + c1 * c1 + c2 * c2;
            if (ds < best) { best = ds; bj = j; b0 = c0; b1 = c1; b2 = c2; }
        }
        mask |= 1u << bj;
        int e = (cry * NPC + i) * KNN + k;
        idxg[e] = cry * NPC + bj;
        dcart[e * 3 + 0] = b0; dcart[e * 3 + 1] = b1; dcart[e * 3 + 2] = b2;
        dsqg[e] = best;
    }
}

// node embedding: h = z[16384,128] @ emb_w[128,64] + b  (WMMA, 64 nodes per block)
__global__ __launch_bounds__(256) void k_embed(const float* __restrict__ z,
                                               const _Float16* __restrict__ wt,
                                               const float* __restrict__ bias,
                                               float* __restrict__ hf,
                                               _Float16* __restrict__ h16) {
    __shared__ __align__(16) _Float16 sZ[64 * 128];
    int tid = threadIdx.x, wave = tid >> 5, lane = tid & 31;
    int node0 = blockIdx.x * 64;
    for (int it = 0; it < 32; ++it) {
        int fi = tid + it * 256;
        sZ[fi] = (_Float16)z[node0 * 128 + fi];
    }
    __syncthreads();
    for (int t = wave; t < 16; t += 8) {
        int row0 = (t >> 2) << 4, n0 = (t & 3) << 4;
        v8f acc = {};
#pragma unroll
        for (int ks = 0; ks < 4; ++ks)
            acc = wmma32(ldA(sZ, row0, 128, ks, lane), ldB(wt, 128, n0, ks * 32, lane), acc);
        int n = lane & 15, mo = (lane & 16) ? 8 : 0;
        float bv = bias[n0 + n];
#pragma unroll
        for (int r = 0; r < 8; ++r) {
            int idx = (node0 + row0 + mo + r) * 64 + n0 + n;
            float v = acc[r] + bv;
            hf[idx] = v; h16[idx] = (_Float16)v;
        }
    }
}

// edge MLP + coord weight + m_i aggregation: 8 dst nodes (96 edges) per block
__global__ __launch_bounds__(256) void k_edge(int l, const _Float16* __restrict__ h16,
        const float* __restrict__ dsqg, const float* __restrict__ dcart,
        const int* __restrict__ idxg, const float* __restrict__ tpe,
        const float* __restrict__ ew1, const _Float16* __restrict__ wt_e1,
        const _Float16* __restrict__ wt_e2, const _Float16* __restrict__ wt_c1,
        const float* __restrict__ eb2, const float* __restrict__ cb1,
        const float* __restrict__ cw2, _Float16* __restrict__ mi_g,
        float* __restrict__ tshift) {
    __shared__ __align__(16) _Float16 sHs[96 * 64];   // h[src] per edge
    __shared__ __align__(16) _Float16 sHd[96 * 64];   // h[dst] per edge
    __shared__ __align__(16) _Float16 sM1[96 * 64];
    __shared__ __align__(16) _Float16 sM2[96 * 64];
    __shared__ int   sidx[96];
    __shared__ float sdist[96];
    __shared__ float sdc[96 * 3];
    __shared__ float stv[64];
    __shared__ float sw1d[64];
    __shared__ float scw[96];
    int tid = threadIdx.x, wave = tid >> 5, lane = tid & 31;
    int node0 = blockIdx.x * 8;
    int cry = node0 >> 5;
    if (tid < 96) {
        int e = node0 * KNN + tid;
        sidx[tid] = idxg[e];
        sdist[tid] = dsqg[e];
        sdc[tid * 3 + 0] = dcart[e * 3 + 0];
        sdc[tid * 3 + 1] = dcart[e * 3 + 1];
        sdc[tid * 3 + 2] = dcart[e * 3 + 2];
    }
    if (tid < 64) {
        stv[tid]  = tpe[(l * BB + cry) * 64 + tid];
        sw1d[tid] = ew1[(l * 193 + 128) * 64 + tid];   // dist row of edge_w1
    }
    __syncthreads();
    // gather h tiles straight into LDS: 96 rows x 128B = 768 x 16B chunks, async DMA
    {
        unsigned lhs = lds_off_of(sHs), lhd = lds_off_of(sHd);
#pragma unroll
        for (int it = 0; it < 3; ++it) {
            int ch = tid + it * 256;                // chunk id 0..767
            int e = ch >> 3, cb = (ch & 7) * 16;    // edge row, byte within row
            const char* gs = (const char*)(h16 + sidx[e] * 64) + cb;
            const char* gd = (const char*)(h16 + (node0 + e / KNN) * 64) + cb;
            async_g2l_b128(lhs + ch * 16, gs);
            async_g2l_b128(lhd + ch * 16, gd);
        }
    }
    wait_async0();
    __syncthreads();
    // MLP1: h_src@W1a + h_dst@W1b + dist*w1d + tproj  -> silu -> sM1
    const _Float16* w1t = wt_e1 + l * 64 * 224;
    for (int t = wave; t < 24; t += 8) {
        int row0 = (t >> 2) << 4, n0 = (t & 3) << 4;
        v8f acc = {};
#pragma unroll
        for (int ks = 0; ks < 2; ++ks)
            acc = wmma32(ldA(sHs, row0, 64, ks, lane), ldB(w1t, 224, n0, ks * 32, lane), acc);
#pragma unroll
        for (int ks = 0; ks < 2; ++ks)
            acc = wmma32(ldA(sHd, row0, 64, ks, lane), ldB(w1t, 224, n0, 64 + ks * 32, lane), acc);
        int n = lane & 15, mo = (lane & 16) ? 8 : 0;
        float tv = stv[n0 + n], wd = sw1d[n0 + n];
#pragma unroll
        for (int r = 0; r < 8; ++r) {
            int m = row0 + mo + r;
            sM1[m * 64 + n0 + n] = (_Float16)silu_f(acc[r] + tv + sdist[m] * wd);
        }
    }
    __syncthreads();
    gemm_silu_96(sM1, sM2, wt_e2 + l * 64 * 64, eb2 + l * 64, wave, lane);  // m
    __syncthreads();
    gemm_silu_96(sM2, sM1, wt_c1 + l * 64 * 64, cb1 + l * 64, wave, lane);  // p
    __syncthreads();
    if (tid < 96) {                                  // cw = p . coord_w2
        const float* w = cw2 + l * 64;
        float s = 0.f;
        for (int c = 0; c < 64; ++c) s += (float)sM1[tid * 64 + c] * w[c];
        scw[tid] = s;
    }
    __syncthreads();
#pragma unroll
    for (int it = 0; it < 2; ++it) {                 // m_i: sum 12 edges per dst node
        int fi = tid + it * 256;
        int ni = fi >> 6, c = fi & 63;
        float s = 0.f;
#pragma unroll
        for (int k = 0; k < KNN; ++k) s += (float)sM2[(ni * KNN + k) * 64 + c];
        mi_g[(node0 + ni) * 64 + c] = (_Float16)s;
    }
    if (tid < 96) {                                  // coord scatter over src
        float w = scw[tid]; int src = sidx[tid];
        atomicAdd(&tshift[src * 3 + 0], sdc[tid * 3 + 0] * w);
        atomicAdd(&tshift[src * 3 + 1], sdc[tid * 3 + 1] * w);
        atomicAdd(&tshift[src * 3 + 2], sdc[tid * 3 + 2] * w);
    }
}

// node MLP + residual: 64 nodes per block
__global__ __launch_bounds__(256) void k_node(int l, float* __restrict__ hf,
        _Float16* __restrict__ h16, const _Float16* __restrict__ mi_g,
        const float* __restrict__ tpn, const _Float16* __restrict__ wt_n1,
        const _Float16* __restrict__ wt_n2, const float* __restrict__ nb2) {
    __shared__ __align__(16) _Float16 sH[64 * 64];
    __shared__ __align__(16) _Float16 sMi[64 * 64];
    __shared__ __align__(16) _Float16 sMid[64 * 64];
    __shared__ float stv[128];                      // 2 crystals per block
    int tid = threadIdx.x, wave = tid >> 5, lane = tid & 31;
    int node0 = blockIdx.x * 64, cry0 = node0 >> 5;
    {   // async DMA both contiguous 8KB tiles into LDS: 512 x 16B chunks each
        unsigned lh = lds_off_of(sH), lm = lds_off_of(sMi);
        const char* gh = (const char*)(h16 + node0 * 64);
        const char* gm = (const char*)(mi_g + node0 * 64);
#pragma unroll
        for (int it = 0; it < 2; ++it) {
            int ch = tid + it * 256;
            async_g2l_b128(lh + ch * 16, gh + ch * 16);
            async_g2l_b128(lm + ch * 16, gm + ch * 16);
        }
    }
    if (tid < 128) stv[tid] = tpn[(l * BB + cry0 + tid / 64) * 64 + (tid & 63)];
    wait_async0();
    __syncthreads();
    const _Float16* w1 = wt_n1 + l * 64 * 192;
    for (int t = wave; t < 16; t += 8) {
        int row0 = (t >> 2) << 4, n0 = (t & 3) << 4;
        v8f acc = {};
#pragma unroll
        for (int ks = 0; ks < 2; ++ks)
            acc = wmma32(ldA(sH, row0, 64, ks, lane), ldB(w1, 192, n0, ks * 32, lane), acc);
#pragma unroll
        for (int ks = 0; ks < 2; ++ks)
            acc = wmma32(ldA(sMi, row0, 64, ks, lane), ldB(w1, 192, n0, 64 + ks * 32, lane), acc);
        int n = lane & 15, mo = (lane & 16) ? 8 : 0;
#pragma unroll
        for (int r = 0; r < 8; ++r) {
            int m = row0 + mo + r;
            sMid[m * 64 + n0 + n] = (_Float16)silu_f(acc[r] + stv[(m >= 32 ? 64 : 0) + n0 + n]);
        }
    }
    __syncthreads();
    const _Float16* w2 = wt_n2 + l * 64 * 64;
    for (int t = wave; t < 16; t += 8) {
        int row0 = (t >> 2) << 4, n0 = (t & 3) << 4;
        v8f acc = {};
#pragma unroll
        for (int ks = 0; ks < 2; ++ks)
            acc = wmma32(ldA(sMid, row0, 64, ks, lane), ldB(w2, 64, n0, ks * 32, lane), acc);
        int n = lane & 15, mo = (lane & 16) ? 8 : 0;
        float bv = nb2[l * 64 + n0 + n];
#pragma unroll
        for (int r = 0; r < 8; ++r) {
            int idx = (node0 + row0 + mo + r) * 64 + n0 + n;
            float v = hf[idx] + acc[r] + bv;        // residual
            hf[idx] = v; h16[idx] = (_Float16)v;
        }
    }
}

__global__ void k_invlat(const float* __restrict__ lat, float* __restrict__ inv) {
    int b = blockIdx.x * blockDim.x + threadIdx.x;
    if (b >= BB) return;
    const float* M = lat + b * 9;
    float a = M[0], bb_ = M[1], c = M[2], d = M[3], e = M[4], f = M[5], g = M[6], h = M[7], i = M[8];
    float det = a * (e * i - f * h) - bb_ * (d * i - f * g) + c * (d * h - e * g);
    float r = 1.f / det;
    float* o = inv + b * 9;
    o[0] = (e * i - f * h) * r; o[1] = (c * h - bb_ * i) * r; o[2] = (bb_ * f - c * e) * r;
    o[3] = (f * g - d * i) * r; o[4] = (a * i - c * g) * r;   o[5] = (c * d - a * f) * r;
    o[6] = (d * h - e * g) * r; o[7] = (bb_ * g - a * h) * r; o[8] = (a * e - bb_ * d) * r;
}

__global__ void k_shift(const float* __restrict__ ts, const float* __restrict__ inv,
                        float* __restrict__ out) {
    int n = blockIdx.x * blockDim.x + threadIdx.x;
    if (n >= NN) return;
    const float* iv = inv + (n >> 5) * 9;
    float t0 = ts[n * 3 + 0], t1 = ts[n * 3 + 1], t2 = ts[n * 3 + 2];
#pragma unroll
    for (int j = 0; j < 3; ++j)
        out[n * 3 + j] = t0 * iv[0 * 3 + j] + t1 * iv[1 * 3 + j] + t2 * iv[2 * 3 + j];
}

__global__ void k_copyh(const float* __restrict__ hf, float* __restrict__ out) {
    int i = blockIdx.x * blockDim.x + threadIdx.x;
    if (i < NN * DD) out[NN * 3 + i] = hf[i];
}

// ---------------- host launcher ----------------
static size_t alignup(size_t x) { return (x + 255) & ~(size_t)255; }

extern "C" void kernel_launch(void* const* d_in, const int* in_sizes, int n_in,
                              void* d_out, int out_size, void* d_ws, size_t ws_size,
                              hipStream_t stream) {
    const float* z       = (const float*)d_in[0];
    const float* t_in    = (const float*)d_in[1];
    const float* frac    = (const float*)d_in[2];
    const float* lat     = (const float*)d_in[3];
    const float* tw1     = (const float*)d_in[6];
    const float* tb1     = (const float*)d_in[7];
    const float* tw2     = (const float*)d_in[8];
    const float* tb2     = (const float*)d_in[9];
    const float* embw    = (const float*)d_in[10];
    const float* embb    = (const float*)d_in[11];
    const float* ew1     = (const float*)d_in[12];
    const float* eb1     = (const float*)d_in[13];
    const float* ew2     = (const float*)d_in[14];
    const float* eb2     = (const float*)d_in[15];
    const float* cw1     = (const float*)d_in[16];
    const float* cb1     = (const float*)d_in[17];
    const float* cw2     = (const float*)d_in[18];
    const float* nw1     = (const float*)d_in[19];
    const float* nb1     = (const float*)d_in[20];
    const float* nw2     = (const float*)d_in[21];
    const float* nb2     = (const float*)d_in[22];
    float* out = (float*)d_out;

    char* ws = (char*)d_ws;
    size_t off = 0;
    auto take = [&](size_t bytes) { size_t o = off; off = alignup(off + bytes); return o; };
    float*    tfeat = (float*)(ws + take(BB * 64 * 4));
    float*    tpe   = (float*)(ws + take(LL * BB * 64 * 4));
    float*    tpn   = (float*)(ws + take(LL * BB * 64 * 4));
    float*    hf    = (float*)(ws + take((size_t)NN * 64 * 4));
    _Float16* h16   = (_Float16*)(ws + take((size_t)NN * 64 * 2));
    _Float16* mi_g  = (_Float16*)(ws + take((size_t)NN * 64 * 2));
    float*    tshift= (float*)(ws + take((size_t)NN * 3 * 4));
    int*      idxg  = (int*)(ws + take((size_t)EE * 4));
    float*    dcart = (float*)(ws + take((size_t)EE * 3 * 4));
    float*    dsqg  = (float*)(ws + take((size_t)EE * 4));
    float*    invl  = (float*)(ws + take(BB * 9 * 4));
    _Float16* wtEmb = (_Float16*)(ws + take(64 * 128 * 2));
    _Float16* wtE1  = (_Float16*)(ws + take(LL * 64 * 224 * 2));
    _Float16* wtE2  = (_Float16*)(ws + take(LL * 64 * 64 * 2));
    _Float16* wtC1  = (_Float16*)(ws + take(LL * 64 * 64 * 2));
    _Float16* wtN1  = (_Float16*)(ws + take(LL * 64 * 192 * 2));
    _Float16* wtN2  = (_Float16*)(ws + take(LL * 64 * 64 * 2));

    auto cvt = [&](const float* s, _Float16* d, int L_, int Kr, int KrPad) {
        int tot = L_ * 64 * KrPad;
        k_f16t<<<(tot + 255) / 256, 256, 0, stream>>>(s, d, L_, Kr, KrPad);
    };
    cvt(embw, wtEmb, 1, 128, 128);
    cvt(ew1,  wtE1, LL, 193, 224);
    cvt(ew2,  wtE2, LL, 64, 64);
    cvt(cw1,  wtC1, LL, 64, 64);
    cvt(nw1,  wtN1, LL, 192, 192);
    cvt(nw2,  wtN2, LL, 64, 64);

    k_time<<<BB, 64, 0, stream>>>(t_in, tw1, tb1, tw2, tb2, tfeat);
    k_tproj<<<LL * BB, 64, 0, stream>>>(tfeat, ew1, eb1, nw1, nb1, tpe, tpn);
    k_knn<<<BB, 32, 0, stream>>>(frac, lat, idxg, dcart, dsqg);
    k_embed<<<NN / 64, 256, 0, stream>>>(z, wtEmb, embb, hf, h16);
    k_zero<<<(NN * 3 + 255) / 256, 256, 0, stream>>>(tshift, NN * 3);

    for (int l = 0; l < LL; ++l) {
        k_edge<<<NN / 8, 256, 0, stream>>>(l, h16, dsqg, dcart, idxg, tpe, ew1,
                                           wtE1, wtE2, wtC1, eb2, cb1, cw2, mi_g, tshift);
        k_node<<<NN / 64, 256, 0, stream>>>(l, hf, h16, mi_g, tpn, wtN1, wtN2, nb2);
    }

    k_invlat<<<(BB + 255) / 256, 256, 0, stream>>>(lat, invl);
    k_shift<<<(NN + 255) / 256, 256, 0, stream>>>(tshift, invl, out);
    k_copyh<<<(NN * DD + 255) / 256, 256, 0, stream>>>(hf, out);
    (void)in_sizes; (void)n_in; (void)out_size; (void)ws_size;
}
